// Money_former_block_28475633172570
// MI455X (gfx1250) — compile-verified, matrix operand order checked by reading, not run on hardware
//
#include <hip/hip_runtime.h>

typedef __attribute__((ext_vector_type(16))) _Float16 v16h;
typedef __attribute__((ext_vector_type(8)))  float    v8f;
typedef __attribute__((ext_vector_type(4)))  unsigned int v4u;
typedef __attribute__((ext_vector_type(8)))  int v8i;
typedef __attribute__((ext_vector_type(4)))  int v4i;

#define LL 2048
#define DM 1024
#define NH 16
#define H2N 32
#define SCALING 0.14433756729740643f   /* 48^-0.5 */
#define EPS_F 1.1920928955078125e-07f

// ---------------- WMMA helpers ----------------

__device__ __forceinline__ v8f wmma_f16(v16h a, v16h b, v8f c) {
  return __builtin_amdgcn_wmma_f32_16x16x32_f16(false, a, false, b, (short)0, c, false, false);
}

template<int STRIDE>
__device__ __forceinline__ v16h ld_frag(const _Float16* tile, int row0, int col0, int lane) {
  // CDNA5 16-bit A/B fragment: lane<16 -> row=lane, K {0..7} and {16..23};
  // lane>=16 -> row=lane-16, K {8..15} and {24..31}. col0 selects 32-wide K chunk.
  int fr = lane & 15;
  int k0 = ((lane >> 4) << 3) + col0;
  const _Float16* p = tile + (row0 + fr) * STRIDE + k0;
  union { float4 f[2]; v16h v; } u;
  u.f[0] = *(const float4*)(p);
  u.f[1] = *(const float4*)(p + 16);
  return u.v;
}

// ---------------- Tensor Data Mover: 64-row x 64-elem f16 tile -> LDS [64][72] ----------
// D# per CDNA5 ISA #8. Pad feature emits 8 halves (4 DWORDs) of pad
// after every 64 halves (32 DWORDs) so the LDS tile lands with a 72-half row stride.

__device__ __forceinline__ void tdm_load_tile(unsigned lds, const _Float16* g,
                                              unsigned stride_elems) {
  unsigned long long ga = (unsigned long long)(uintptr_t)g;
  v4u g0 = { 1u,                                   // count=1 (user descriptor)
             lds,                                  // lds_addr (bytes)
             (unsigned)(ga & 0xffffffffu),         // global_addr[31:0]
             (unsigned)((ga >> 32) & 0x01ffffffu) | 0x80000000u };  // addr[56:32] | type=2
  v8i g1;
  g1[0] = (int)((1u << 16)      // data_size = 2 bytes
              | (1u << 20)      // pad_enable
              | (4u << 22)      // pad_interval: 32 DWORDs
              | (3u << 25));    // pad_amount: 4 DWORDs
  g1[1] = (int)((stride_elems & 0xffffu) << 16);               // tensor_dim0 lo16
  g1[2] = (int)(((stride_elems >> 16) & 0xffffu) | (64u << 16)); // tensor_dim0 hi | tensor_dim1 lo
  g1[3] = (int)(64u << 16);                                    // tensor_dim1 hi=0 | tile_dim0=64
  g1[4] = 64;                                                  // tile_dim1=64, tile_dim2=0
  g1[5] = (int)stride_elems;                                   // tensor_dim0_stride lo32
  g1[6] = 0;
  g1[7] = 0;
  v4i z4 = { 0, 0, 0, 0 };
  v8i z8 = { 0, 0, 0, 0, 0, 0, 0, 0 };
  __builtin_amdgcn_tensor_load_to_lds(g0, g1, z4, z4, z8, 0);
}

__device__ __forceinline__ unsigned lds_off(const void* p) {
  return (unsigned)(uintptr_t)p;  // flat LDS address: offset lives in addr[31:0]
}

// compute 64x64 score tile = Q(64x64) * Kt(64x64)^T into S (fp32 LDS), one wave quadrant each
__device__ __forceinline__ void score_tile(_Float16 (*Q)[72], _Float16 (*Kt)[72],
                                           float (*S)[65], int wave, int lane) {
  int wm = (wave >> 1) * 32, wn = (wave & 1) * 32;
  v8f sc[2][2] = {};
  for (int kk = 0; kk < 64; kk += 32) {
    v16h a0 = ld_frag<72>(&Q[0][0],  wm,      kk, lane);
    v16h a1 = ld_frag<72>(&Q[0][0],  wm + 16, kk, lane);
    v16h b0 = ld_frag<72>(&Kt[0][0], wn,      kk, lane);
    v16h b1 = ld_frag<72>(&Kt[0][0], wn + 16, kk, lane);
    sc[0][0] = wmma_f16(a0, b0, sc[0][0]);
    sc[0][1] = wmma_f16(a0, b1, sc[0][1]);
    sc[1][0] = wmma_f16(a1, b0, sc[1][0]);
    sc[1][1] = wmma_f16(a1, b1, sc[1][1]);
  }
  int r0 = (lane >> 4) << 3, c0 = lane & 15;
  for (int mi = 0; mi < 2; ++mi)
    for (int ni = 0; ni < 2; ++ni)
      for (int rr = 0; rr < 8; ++rr)
        S[wm + mi * 16 + r0 + rr][wn + ni * 16 + c0] = sc[mi][ni][rr];
}

// ---------------- generic GEMM: C[M,N] = A[M,K] (f16, row-major) * BT[N,K]^T, fp32 out ----

__global__ __launch_bounds__(128) void gemm_f16(const _Float16* __restrict__ A,
                                                const _Float16* __restrict__ BT,
                                                float* __restrict__ C,
                                                int M, int N, int K) {
  __shared__ _Float16 As[64][40];
  __shared__ _Float16 Bs[64][40];
  const int bm = blockIdx.x * 64, bn = blockIdx.y * 64;
  const int tid = threadIdx.x, lane = tid & 31, wave = tid >> 5;
  const int wm = (wave >> 1) * 32, wn = (wave & 1) * 32;
  v8f acc[2][2] = {};
  const int lr = tid >> 1, lc = (tid & 1) * 16;
  for (int k0 = 0; k0 < K; k0 += 32) {
    const _Float16* pa = A + (size_t)(bm + lr) * K + k0 + lc;
    float4 a0 = *(const float4*)pa;
    float4 a1 = *(const float4*)(pa + 8);
    float4 b0 = make_float4(0.f, 0.f, 0.f, 0.f), b1 = b0;
    if (bn + lr < N) {
      const _Float16* pb = BT + (size_t)(bn + lr) * K + k0 + lc;
      b0 = *(const float4*)pb;
      b1 = *(const float4*)(pb + 8);
      __builtin_prefetch(pb + 32, 0, 1);
    }
    __builtin_prefetch(pa + 32, 0, 1);
    __syncthreads();
    *(float4*)&As[lr][lc]     = a0;
    *(float4*)&As[lr][lc + 8] = a1;
    *(float4*)&Bs[lr][lc]     = b0;
    *(float4*)&Bs[lr][lc + 8] = b1;
    __syncthreads();
    v16h af0 = ld_frag<40>(&As[0][0], wm,      0, lane);
    v16h af1 = ld_frag<40>(&As[0][0], wm + 16, 0, lane);
    v16h bf0 = ld_frag<40>(&Bs[0][0], wn,      0, lane);
    v16h bf1 = ld_frag<40>(&Bs[0][0], wn + 16, 0, lane);
    acc[0][0] = wmma_f16(af0, bf0, acc[0][0]);
    acc[0][1] = wmma_f16(af0, bf1, acc[0][1]);
    acc[1][0] = wmma_f16(af1, bf0, acc[1][0]);
    acc[1][1] = wmma_f16(af1, bf1, acc[1][1]);
  }
  const int r0 = (lane >> 4) << 3, c0 = lane & 15;
  for (int mi = 0; mi < 2; ++mi)
    for (int ni = 0; ni < 2; ++ni) {
      int cc = bn + wn + ni * 16 + c0;
      if (cc < N)
        for (int rr = 0; rr < 8; ++rr)
          C[(size_t)(bm + wm + mi * 16 + r0 + rr) * N + cc] = acc[mi][ni][rr];
    }
}

// ---------------- elementwise / norm kernels ----------------

__global__ __launch_bounds__(256) void transpose_w(const float* __restrict__ W,
                                                   _Float16* __restrict__ WT, int K, int N) {
  __shared__ float tile[32][33];
  int n0 = blockIdx.x * 32, k0 = blockIdx.y * 32;
  int tx = threadIdx.x & 31, ty = threadIdx.x >> 5;
  for (int j = 0; j < 32; j += 8)
    tile[ty + j][tx] = W[(size_t)(k0 + ty + j) * N + n0 + tx];
  __syncthreads();
  for (int j = 0; j < 32; j += 8)
    WT[(size_t)(n0 + ty + j) * K + k0 + tx] = (_Float16)tile[tx][ty + j];
}

__global__ __launch_bounds__(256) void rmsnorm_to_f16(const float* __restrict__ in, int instride,
                                                      int cols, const float* __restrict__ w,
                                                      _Float16* __restrict__ out, float eps) {
  const float* p = in + (size_t)blockIdx.x * instride;
  float ss = 0.f;
  for (int c = threadIdx.x; c < cols; c += 256) { float v = p[c]; ss += v * v; }
  __shared__ float red[256];
  red[threadIdx.x] = ss;
  __syncthreads();
  for (int s = 128; s; s >>= 1) { if (threadIdx.x < s) red[threadIdx.x] += red[threadIdx.x + s]; __syncthreads(); }
  float sc = rsqrtf(red[0] / (float)cols + eps);
  _Float16* o = out + (size_t)blockIdx.x * cols;
  for (int c = threadIdx.x; c < cols; c += 256) o[c] = (_Float16)(p[c] * sc * w[c]);
}

__global__ __launch_bounds__(256) void add_f32(const float* __restrict__ a,
                                               const float* __restrict__ b,
                                               float* __restrict__ o, int n) {
  int i = blockIdx.x * 256 + threadIdx.x;
  if (i < n) o[i] = a[i] + b[i];
}

__global__ __launch_bounds__(256) void zero_f32(float* __restrict__ p, int n) {
  int i = blockIdx.x * 256 + threadIdx.x;
  if (i < n) p[i] = 0.f;
}

__global__ __launch_bounds__(256) void swiglu_f16(const float* __restrict__ uv,
                                                  _Float16* __restrict__ t16) {
  int i = blockIdx.x * 256 + threadIdx.x;   // over 2048*4096
  int r = i >> 12, c = i & 4095;
  float u = uv[(size_t)r * 8192 + c];
  float g = uv[(size_t)r * 8192 + 4096 + c];
  t16[i] = (_Float16)(u * g / (1.f + expf(-g)));
}

__global__ void lam_kernel(const float* __restrict__ lq1, const float* __restrict__ lk1,
                           const float* __restrict__ lq2, const float* __restrict__ lk2,
                           float* __restrict__ lam) {
  int t = threadIdx.x;  // 32 threads
  float v1 = lq1[t] * lk1[t];
  float v2 = lq2[t] * lk2[t];
  for (int off = 16; off; off >>= 1) {
    v1 += __shfl_down(v1, off);
    v2 += __shfl_down(v2, off);
  }
  if (t == 0) *lam = expf(v1) - expf(v2) + 0.2f;  // LAMBDA_INIT = 0.2
}

__global__ __launch_bounds__(64) void head_norm(const float* __restrict__ Obuf,
                                                const float* __restrict__ w,
                                                _Float16* __restrict__ out) {
  int t = blockIdx.x, h = blockIdx.y, d = threadIdx.x;
  float v = Obuf[((size_t)h * LL + t) * 64 + d];
  __shared__ float red[64];
  red[d] = v * v;
  __syncthreads();
  for (int s = 32; s; s >>= 1) { if (d < s) red[d] += red[d + s]; __syncthreads(); }
  float sc = rsqrtf(red[0] * (1.f / 64.f) + 1e-5f);
  out[(size_t)t * 1024 + h * 64 + d] = (_Float16)(v * sc * w[d]);
}

// ---------------- q/k/v assembly with RoPE (q pre-scaled by SCALING) ----------------

__global__ __launch_bounds__(256) void qkv_assemble(const float* __restrict__ qfull,
                                                    const float* __restrict__ kvbuf,
                                                    const float* __restrict__ ckv,
                                                    const float* __restrict__ frq,
                                                    const float* __restrict__ fiq,
                                                    _Float16* __restrict__ qh,
                                                    _Float16* __restrict__ kh,
                                                    _Float16* __restrict__ vT) {
  const int t = blockIdx.x, tid = threadIdx.x;
  const int r = t & 511;
  for (int idx = tid; idx < 1024; idx += 256) {
    int h2 = idx >> 5, d = idx & 31, n = h2 >> 1, e = h2 & 1;
    size_t ob = ((size_t)h2 * LL + t) * 64;
    qh[ob + d] = (_Float16)(qfull[(size_t)t * 1536 + n * 96 + e * 32 + d] * SCALING);
    kh[ob + d] = (_Float16)(kvbuf[(size_t)t * 2048 + n * 128 + e * 32 + d]);
  }
  {
    int idx = tid;  // exactly 256 rope pairs: 32 sub-heads x 8 pairs
    int h2 = idx >> 3, i = idx & 7, n = h2 >> 1, e = h2 & 1;
    float cr = 1.f, ci = 0.f;
    if (r > 0) { cr = frq[(r - 1) * 8 + i]; ci = fiq[(r - 1) * 8 + i]; }
    size_t ob = ((size_t)h2 * LL + t) * 64 + 32;
    float a = qfull[(size_t)t * 1536 + n * 96 + 64 + e * 16 + 2 * i];
    float b = qfull[(size_t)t * 1536 + n * 96 + 64 + e * 16 + 2 * i + 1];
    qh[ob + 2 * i]     = (_Float16)((a * cr - b * ci) * SCALING);
    qh[ob + 2 * i + 1] = (_Float16)((a * ci + b * cr) * SCALING);
    float ka = ckv[(size_t)t * 544 + 512 + e * 16 + 2 * i];
    float kb = ckv[(size_t)t * 544 + 512 + e * 16 + 2 * i + 1];
    kh[ob + 2 * i]     = (_Float16)(ka * cr - kb * ci);
    kh[ob + 2 * i + 1] = (_Float16)(ka * ci + kb * cr);
  }
  for (int idx = tid; idx < 512; idx += 256) {  // zero-pad dims 48..63
    int h2 = idx >> 4, d = idx & 15;
    size_t ob = ((size_t)h2 * LL + t) * 64 + 48;
    qh[ob + d] = (_Float16)0.f;
    kh[ob + d] = (_Float16)0.f;
  }
  for (int idx = tid; idx < 1024; idx += 256) {  // v transposed: vT[n][d][t]
    int n = idx >> 6, d = idx & 63;
    vT[((size_t)n * 64 + d) * LL + t] = (_Float16)(kvbuf[(size_t)t * 2048 + n * 128 + 64 + d]);
  }
}

// ---------------- attention pass 1: per-row max / sum of exp(|s|) ----------------

__global__ __launch_bounds__(128) void attn_stats(const _Float16* __restrict__ qh,
                                                  const _Float16* __restrict__ kh,
                                                  float* __restrict__ rowmax,
                                                  float* __restrict__ rowsum) {
  __shared__ _Float16 Qs[64][72];
  __shared__ _Float16 Ks[64][72];
  __shared__ float S[64][65];
  const int qt = blockIdx.x, h2 = blockIdx.y;
  const int tid = threadIdx.x, lane = tid & 31, wave = tid >> 5;
  if (tid == 0)
    tdm_load_tile(lds_off(&Qs[0][0]), qh + ((size_t)h2 * LL + qt * 64) * 64, 64);
  float m_run = -1e30f, s_run = 0.f;
  const int qg = qt * 64 + tid;
  const int qm = qg & 511;
  for (int kt = 0; kt < 32; ++kt) {
    __syncthreads();
    if (tid == 0)
      tdm_load_tile(lds_off(&Ks[0][0]), kh + ((size_t)h2 * LL + kt * 64) * 64, 64);
    __builtin_amdgcn_s_wait_tensorcnt(0);
    __syncthreads();
    score_tile(Qs, Ks, S, wave, lane);
    __syncthreads();
    if (tid < 64) {
      float tm = -1e30f;
      for (int c = 0; c < 64; ++c)
        if (((kt * 64 + c) & 511) <= qm) tm = fmaxf(tm, fabsf(S[tid][c]));
      float mn = fmaxf(m_run, tm);
      float ssum = 0.f;
      for (int c = 0; c < 64; ++c)
        if (((kt * 64 + c) & 511) <= qm) ssum += expf(fabsf(S[tid][c]) - mn);
      s_run = s_run * expf(m_run - mn) + ssum;
      m_run = mn;
    }
  }
  if (tid < 64) {
    rowmax[(size_t)h2 * LL + qg] = m_run;
    rowsum[(size_t)h2 * LL + qg] = s_run;
  }
}

// ---------------- attention pass 2: column sums of p1 (even sub-heads) ----------------

__global__ __launch_bounds__(128) void attn_colsum(const _Float16* __restrict__ qh,
                                                   const _Float16* __restrict__ kh,
                                                   const float* __restrict__ rowmax,
                                                   const float* __restrict__ rowsum,
                                                   float* __restrict__ gcol) {
  __shared__ _Float16 Qs[64][72];
  __shared__ _Float16 Ks[64][72];
  __shared__ float S[64][65];
  __shared__ float rmv[64], rsv[64];
  const int qt = blockIdx.x, nh = blockIdx.y, h2 = nh * 2;
  const int tid = threadIdx.x, lane = tid & 31, wave = tid >> 5;
  if (tid == 0)
    tdm_load_tile(lds_off(&Qs[0][0]), qh + ((size_t)h2 * LL + qt * 64) * 64, 64);
  if (tid < 64) {
    rmv[tid] = rowmax[(size_t)h2 * LL + qt * 64 + tid];
    rsv[tid] = rowsum[(size_t)h2 * LL + qt * 64 + tid];
  }
  for (int kt = 0; kt < 32; ++kt) {
    __syncthreads();
    if (tid == 0)
      tdm_load_tile(lds_off(&Ks[0][0]), kh + ((size_t)h2 * LL + kt * 64) * 64, 64);
    __builtin_amdgcn_s_wait_tensorcnt(0);
    __syncthreads();
    score_tile(Qs, Ks, S, wave, lane);
    __syncthreads();
    if (tid < 64) {
      int key = kt * 64 + tid, km = key & 511;
      float cs = 0.f;
      for (int q2 = 0; q2 < 64; ++q2)
        if (km <= ((qt * 64 + q2) & 511)) {
          float s = S[q2][tid];
          cs += copysignf(expf(fabsf(s) - rmv[q2]) / rsv[q2], s);
        }
      atomicAdd(&gcol[(size_t)nh * LL + key], cs);
    }
  }
}

// ---------------- attention pass 3: O = (p1 - lam*p2 + lam*gmean) @ V ----------------

__global__ __launch_bounds__(128) void attn_out(const _Float16* __restrict__ qh,
                                                const _Float16* __restrict__ kh,
                                                const _Float16* __restrict__ vT,
                                                const float* __restrict__ rowmax,
                                                const float* __restrict__ rowsum,
                                                const float* __restrict__ gcol,
                                                const float* __restrict__ lamp,
                                                float* __restrict__ Obuf) {
  __shared__ _Float16 Qa[64][72], Qb[64][72], Ks[64][72], Vts[64][72], Ps[64][72];
  __shared__ float S[64][65];
  __shared__ float rma[64], rsa[64], rmb[64], rsb[64];
  const int qt = blockIdx.x, nh = blockIdx.y;
  const int h2a = nh * 2, h2b = h2a + 1;
  const int tid = threadIdx.x, lane = tid & 31, wave = tid >> 5;
  const int wm = (wave >> 1) * 32, wn = (wave & 1) * 32;
  const float lam = *lamp;
  if (tid == 0) {
    tdm_load_tile(lds_off(&Qa[0][0]), qh + ((size_t)h2a * LL + qt * 64) * 64, 64);
    tdm_load_tile(lds_off(&Qb[0][0]), qh + ((size_t)h2b * LL + qt * 64) * 64, 64);
  }
  if (tid < 64) {
    rma[tid] = rowmax[(size_t)h2a * LL + qt * 64 + tid];
    rsa[tid] = rowsum[(size_t)h2a * LL + qt * 64 + tid];
    rmb[tid] = rowmax[(size_t)h2b * LL + qt * 64 + tid];
    rsb[tid] = rowsum[(size_t)h2b * LL + qt * 64 + tid];
  }
  v8f oacc[2][2] = {};
  for (int kt = 0; kt < 32; ++kt) {
    __syncthreads();
    if (tid == 0)
      tdm_load_tile(lds_off(&Ks[0][0]), kh + ((size_t)h2a * LL + kt * 64) * 64, 64);
    __builtin_amdgcn_s_wait_tensorcnt(0);
    __syncthreads();
    score_tile(Qa, Ks, S, wave, lane);
    __syncthreads();
    for (int idx = tid; idx < 4096; idx += 128) {
      int q2 = idx >> 6, c2 = idx & 63;
      int key = kt * 64 + c2;
      float p1 = 0.f;
      if ((key & 511) <= ((qt * 64 + q2) & 511)) {
        float s = S[q2][c2];
        p1 = copysignf(expf(fabsf(s) - rma[q2]) / rsa[q2], s);
      }
      Ps[q2][c2] = (_Float16)p1;
    }
    __syncthreads();
    if (tid == 0)
      tdm_load_tile(lds_off(&Ks[0][0]), kh + ((size_t)h2b * LL + kt * 64) * 64, 64);
    __builtin_amdgcn_s_wait_tensorcnt(0);
    __syncthreads();
    score_tile(Qb, Ks, S, wave, lane);
    __syncthreads();
    for (int idx = tid; idx < 4096; idx += 128) {
      int q2 = idx >> 6, c2 = idx & 63;
      int key = kt * 64 + c2;
      float pv = 0.f;
      if ((key & 511) <= ((qt * 64 + q2) & 511)) {
        float s = S[q2][c2];
        float p2 = copysignf(expf(fabsf(s) - rmb[q2]) / rsb[q2], s);
        pv = (float)Ps[q2][c2] - lam * p2 + lam * (gcol[(size_t)nh * LL + key] * (1.f / 2048.f));
      }
      Ps[q2][c2] = (_Float16)pv;
    }
    __syncthreads();
    if (tid == 0)
      tdm_load_tile(lds_off(&Vts[0][0]), vT + (size_t)nh * 64 * LL + kt * 64, LL);
    __builtin_amdgcn_s_wait_tensorcnt(0);
    __syncthreads();
    for (int kk = 0; kk < 64; kk += 32) {
      v16h a0 = ld_frag<72>(&Ps[0][0],  wm,      kk, lane);
      v16h a1 = ld_frag<72>(&Ps[0][0],  wm + 16, kk, lane);
      v16h b0 = ld_frag<72>(&Vts[0][0], wn,      kk, lane);
      v16h b1 = ld_frag<72>(&Vts[0][0], wn + 16, kk, lane);
      oacc[0][0] = wmma_f16(a0, b0, oacc[0][0]);
      oacc[0][1] = wmma_f16(a0, b1, oacc[0][1]);
      oacc[1][0] = wmma_f16(a1, b0, oacc[1][0]);
      oacc[1][1] = wmma_f16(a1, b1, oacc[1][1]);
    }
  }
  const int r0 = (lane >> 4) << 3, c0 = lane & 15;
  for (int mi = 0; mi < 2; ++mi)
    for (int ni = 0; ni < 2; ++ni)
      for (int rr = 0; rr < 8; ++rr)
        Obuf[((size_t)nh * LL + qt * 64 + wm + mi * 16 + r0 + rr) * 64 + wn + ni * 16 + c0] =
            oacc[mi][ni][rr];
}

// ---------------- host orchestration ----------------

extern "C" void kernel_launch(void* const* d_in, const int* in_sizes, int n_in,
                              void* d_out, int out_size, void* d_ws, size_t ws_size,
                              hipStream_t stream) {
  const float* x        = (const float*)d_in[0];
  const float* freqs_r  = (const float*)d_in[1];
  const float* freqs_i  = (const float*)d_in[2];
  const float* kv_down  = (const float*)d_in[4];
  const float* q_down   = (const float*)d_in[5];
  const float* kv_up    = (const float*)d_in[6];
  const float* q_up     = (const float*)d_in[7];
  const float* kv_nw    = (const float*)d_in[8];
  const float* q_nw     = (const float*)d_in[9];
  const float* o_w      = (const float*)d_in[10];
  const float* lq1      = (const float*)d_in[11];
  const float* lk1      = (const float*)d_in[12];
  const float* lq2      = (const float*)d_in[13];
  const float* lk2      = (const float*)d_in[14];
  const float* head_w   = (const float*)d_in[15];
  const float* n1w      = (const float*)d_in[16];
  const float* n2w      = (const float*)d_in[17];
  const float* ff_in    = (const float*)d_in[18];
  const float* ff_out   = (const float*)d_in[19];
  float* out = (float*)d_out;

  char* w = (char*)d_ws;
  size_t off = 0;
  auto take = [&](size_t b) { char* p = w + off; off = (off + b + 255) & ~(size_t)255; return p; };

  // persistent
  _Float16* wT_kvd = (_Float16*)take((size_t)544 * 1024 * 2);
  _Float16* wT_qd  = (_Float16*)take((size_t)512 * 1024 * 2);
  _Float16* wT_kvu = (_Float16*)take((size_t)2048 * 512 * 2);
  _Float16* wT_qu  = (_Float16*)take((size_t)1536 * 512 * 2);
  _Float16* wT_o   = (_Float16*)take((size_t)1024 * 1024 * 2);
  _Float16* wT_ffi = (_Float16*)take((size_t)8192 * 1024 * 2);
  _Float16* wT_ffo = (_Float16*)take((size_t)1024 * 4096 * 2);
  float* x2     = (float*)take((size_t)LL * DM * 4);
  float* lamb   = (float*)take(256);
  float* rowmax = (float*)take((size_t)H2N * LL * 4);
  float* rowsum = (float*)take((size_t)H2N * LL * 4);
  float* gcol   = (float*)take((size_t)NH * LL * 4);

  // phase-overlaid scratch
  size_t phase0 = off;
  _Float16* h16   = (_Float16*)take((size_t)LL * DM * 2);
  float* ckv      = (float*)take((size_t)LL * 544 * 4);
  float* cq       = (float*)take((size_t)LL * 512 * 4);
  _Float16* ckvn  = (_Float16*)take((size_t)LL * 512 * 2);
  _Float16* cqn   = (_Float16*)take((size_t)LL * 512 * 2);
  float* kvbuf    = (float*)take((size_t)LL * 2048 * 4);
  float* qfull    = (float*)take((size_t)LL * 1536 * 4);
  _Float16* qhb   = (_Float16*)take((size_t)H2N * LL * 64 * 2);
  _Float16* khb   = (_Float16*)take((size_t)H2N * LL * 64 * 2);
  _Float16* vTb   = (_Float16*)take((size_t)NH * 64 * LL * 2);
  float* Obuf     = (float*)take((size_t)NH * LL * 64 * 4);
  _Float16* aout16 = (_Float16*)take((size_t)LL * DM * 2);
  float* aproj    = (float*)take((size_t)LL * DM * 4);
  // FFN phase reuses the same region (ordering on stream makes this safe)
  off = phase0;
  _Float16* hn2 = (_Float16*)take((size_t)LL * DM * 2);
  float* uv     = (float*)take((size_t)LL * 8192 * 4);
  _Float16* t16 = (_Float16*)take((size_t)LL * 4096 * 2);
  float* ffb    = (float*)take((size_t)LL * DM * 4);

  // 0) weight transposes (fp32 KxN -> f16 NxK)
  transpose_w<<<dim3(17, 32), 256, 0, stream>>>(kv_down, wT_kvd, 1024, 544);
  transpose_w<<<dim3(16, 32), 256, 0, stream>>>(q_down, wT_qd, 1024, 512);
  transpose_w<<<dim3(64, 16), 256, 0, stream>>>(kv_up, wT_kvu, 512, 2048);
  transpose_w<<<dim3(48, 16), 256, 0, stream>>>(q_up, wT_qu, 512, 1536);
  transpose_w<<<dim3(32, 32), 256, 0, stream>>>(o_w, wT_o, 1024, 1024);
  transpose_w<<<dim3(256, 32), 256, 0, stream>>>(ff_in, wT_ffi, 1024, 8192);
  transpose_w<<<dim3(32, 128), 256, 0, stream>>>(ff_out, wT_ffo, 4096, 1024);

  // 1) h = rmsnorm(x, norm1)
  rmsnorm_to_f16<<<LL, 256, 0, stream>>>(x, DM, DM, n1w, h16, EPS_F);
  // 2) down projections
  gemm_f16<<<dim3(32, 9), 128, 0, stream>>>(h16, wT_kvd, ckv, LL, 544, 1024);
  gemm_f16<<<dim3(32, 8), 128, 0, stream>>>(h16, wT_qd, cq, LL, 512, 1024);
  // 3) latent norms
  rmsnorm_to_f16<<<LL, 256, 0, stream>>>(ckv, 544, 512, kv_nw, ckvn, EPS_F);
  rmsnorm_to_f16<<<LL, 256, 0, stream>>>(cq, 512, 512, q_nw, cqn, EPS_F);
  // 4) up projections
  gemm_f16<<<dim3(32, 32), 128, 0, stream>>>(ckvn, wT_kvu, kvbuf, LL, 2048, 512);
  gemm_f16<<<dim3(32, 24), 128, 0, stream>>>(cqn, wT_qu, qfull, LL, 1536, 512);
  // 5) assemble q/k/v with RoPE
  qkv_assemble<<<LL, 256, 0, stream>>>(qfull, kvbuf, ckv, freqs_r, freqs_i, qhb, khb, vTb);
  // 6) lambda + zero colsum accumulator
  lam_kernel<<<1, 32, 0, stream>>>(lq1, lk1, lq2, lk2, lamb);
  zero_f32<<<(NH * LL + 255) / 256, 256, 0, stream>>>(gcol, NH * LL);
  // 7) attention passes
  attn_stats<<<dim3(32, 32), 128, 0, stream>>>(qhb, khb, rowmax, rowsum);
  attn_colsum<<<dim3(32, 16), 128, 0, stream>>>(qhb, khb, rowmax, rowsum, gcol);
  attn_out<<<dim3(32, 16), 128, 0, stream>>>(qhb, khb, vTb, rowmax, rowsum, gcol, lamb, Obuf);
  // 8) head rmsnorm + layout
  head_norm<<<dim3(LL, NH), 64, 0, stream>>>(Obuf, head_w, aout16);
  // 9) output projection + residual
  gemm_f16<<<dim3(32, 16), 128, 0, stream>>>(aout16, wT_o, aproj, LL, 1024, 1024);
  add_f32<<<(LL * DM + 255) / 256, 256, 0, stream>>>(x, aproj, x2, LL * DM);
  // 10) FFN
  rmsnorm_to_f16<<<LL, 256, 0, stream>>>(x2, DM, DM, n2w, hn2, EPS_F);
  gemm_f16<<<dim3(32, 128), 128, 0, stream>>>(hn2, wT_ffi, uv, LL, 8192, 1024);
  swiglu_f16<<<(LL * 4096 + 255) / 256, 256, 0, stream>>>(uv, t16);
  gemm_f16<<<dim3(32, 16), 128, 0, stream>>>(t16, wT_ffo, ffb, LL, 1024, 4096);
  add_f32<<<(LL * DM + 255) / 256, 256, 0, stream>>>(x2, ffb, out, LL * DM);
}
